// SwinTransformerBlock_81690277970129
// MI455X (gfx1250) — compile-verified, matrix operand order checked by reading, not run on hardware
//
#include <hip/hip_runtime.h>
#include <hip/hip_bf16.h>

// ---------------------------------------------------------------------------
// Problem constants (fixed by setup_inputs): B=1, H=W=D=28, N=28^3=21952,
// C=256, HEADS=8 (4 per branch), HD=32, SR=4, WS=7, Nr=343 (pad 352).
// ---------------------------------------------------------------------------
#define NTOK   21952      // 28*28*28
#define NT_M   1372       // NTOK/16
#define NRPAD  352        // 343 padded to 22 tiles of 16
#define SCALE_ 0.17677669529663687f   // 32^-0.5

typedef __attribute__((ext_vector_type(16))) __bf16        v16bf;
typedef __attribute__((ext_vector_type(8)))  float         v8f;
typedef __attribute__((ext_vector_type(4)))  unsigned int  u32x4;

union FragCvt { u32x4 u[2]; v16bf v; };

// Load a 16-element bf16 fragment: two contiguous 16B chunks at p and p+16
// (p already includes the per-lane k-base + hi*8 offset per the CDNA5
// 16-bit A/B VGPR layout: lane holds K = {hi*8..hi*8+7, 16+hi*8..16+hi*8+7}).
__device__ __forceinline__ v16bf ld_frag(const unsigned short* p) {
  FragCvt f;
  f.u[0] = *(const u32x4*)(p);
  f.u[1] = *(const u32x4*)(p + 16);
  return f.v;
}

__device__ __forceinline__ v8f wmma_bf16(v16bf a, v16bf b, v8f c) {
  return __builtin_amdgcn_wmma_f32_16x16x32_bf16(false, a, false, b,
                                                 (short)0, c, false, false);
}

__device__ __forceinline__ unsigned short f2bf(float f) {
  unsigned int u = __float_as_uint(f);
  unsigned int r = (u + 0x7fffu + ((u >> 16) & 1u)) >> 16;
  return (unsigned short)r;
}
__device__ __forceinline__ float bf2f(unsigned short h) {
  return __uint_as_float(((unsigned int)h) << 16);
}

// ---------------------------------------------------------------------------
// K0a: x (fp32 [N,256]) -> bf16
// ---------------------------------------------------------------------------
__global__ void k_convert_x(const float* __restrict__ x,
                            unsigned short* __restrict__ xbf) {
  int i = blockIdx.x * 256 + threadIdx.x;
  if (i < NTOK * 256) xbf[i] = f2bf(x[i]);
}

// ---------------------------------------------------------------------------
// K0b: pack all weights to bf16.
//   wcat [768][256] = rows: 0..255 lepe_w | 256..383 q1_w | 384..511 q2_w |
//                     512..767 kv2_w   (all row-major [out][in])
//   kv1w [256][256], projw [256][256], srw [256][16384]
// ---------------------------------------------------------------------------
__global__ void k_pack_w(const float* __restrict__ lepe_w,
                         const float* __restrict__ q1_w,
                         const float* __restrict__ q2_w,
                         const float* __restrict__ kv2_w,
                         const float* __restrict__ kv1_w,
                         const float* __restrict__ proj_w,
                         const float* __restrict__ sr_w,
                         unsigned short* __restrict__ wcat,
                         unsigned short* __restrict__ kv1bf,
                         unsigned short* __restrict__ projbf,
                         unsigned short* __restrict__ srbf) {
  int i = blockIdx.x * 256 + threadIdx.x;
  const int n_cat = 768 * 256, n_sq = 256 * 256, n_sr = 256 * 16384;
  if (i < n_cat) {
    int orow = i >> 8, col = i & 255;
    float v;
    if (orow < 256)      v = lepe_w[orow * 256 + col];
    else if (orow < 384) v = q1_w[(orow - 256) * 256 + col];
    else if (orow < 512) v = q2_w[(orow - 384) * 256 + col];
    else                 v = kv2_w[(orow - 512) * 256 + col];
    wcat[i] = f2bf(v);
    return;
  }
  i -= n_cat;
  if (i < n_sq) { kv1bf[i] = f2bf(kv1_w[i]); return; }
  i -= n_sq;
  if (i < n_sq) { projbf[i] = f2bf(proj_w[i]); return; }
  i -= n_sq;
  if (i < n_sr) srbf[i] = f2bf(sr_w[i]);
}

// ---------------------------------------------------------------------------
// K1: fused GEMM D[N,768] = xbf @ wcat^T, scatter-stored:
//   cols   0..255 -> y fp32 [N,256] (+lepe_b)     (input to depthwise conv)
//   cols 256..383 -> q1 bf16 [head][N][32]
//   cols 384..511 -> q2w bf16 [head][win][352][32] (window-partitioned)
//   cols 512..639 -> k2w bf16 [head][win][352][32]
//   cols 640..767 -> v2wT bf16 [head][win][32][352] (transposed for PV)
// grid (1372, 12), block 32 (1 wave does 16x64 of D).
// ---------------------------------------------------------------------------
__global__ void k_gemm_fused(const unsigned short* __restrict__ xbf,
                             const unsigned short* __restrict__ wcat,
                             const float* __restrict__ lepe_b,
                             float* __restrict__ y,
                             unsigned short* __restrict__ q1,
                             unsigned short* __restrict__ q2w,
                             unsigned short* __restrict__ k2w,
                             unsigned short* __restrict__ v2wT) {
  const int lane = threadIdx.x & 31;
  const int mn = lane & 15, hi = lane >> 4;
  const int mt = blockIdx.x, ng = blockIdx.y;
  v8f acc[4] = { {0,0,0,0,0,0,0,0}, {0,0,0,0,0,0,0,0},
                 {0,0,0,0,0,0,0,0}, {0,0,0,0,0,0,0,0} };
  const unsigned short* arow = xbf + (size_t)(mt * 16 + mn) * 256;
  for (int ks = 0; ks < 8; ++ks) {
    v16bf a = ld_frag(arow + ks * 32 + hi * 8);
    #pragma unroll
    for (int j = 0; j < 4; ++j) {
      const unsigned short* brow = wcat + (size_t)(ng * 64 + j * 16 + mn) * 256;
      v16bf b = ld_frag(brow + ks * 32 + hi * 8);
      acc[j] = wmma_bf16(a, b, acc[j]);
    }
  }
  #pragma unroll
  for (int j = 0; j < 4; ++j) {
    #pragma unroll
    for (int v = 0; v < 8; ++v) {
      int mm = v + 8 * hi;
      int t  = mt * 16 + mm;
      int gc = ng * 64 + j * 16 + mn;
      float val = acc[j][v];
      if (gc < 256) {
        y[(size_t)t * 256 + gc] = val + lepe_b[gc];
      } else if (gc < 384) {
        int c = gc - 256;
        q1[(size_t)((c >> 5) * NTOK + t) * 32 + (c & 31)] = f2bf(val);
      } else {
        int h = t / 784, w = (t / 28) % 28, d = t % 28;
        int wdw   = (h / 7) * 16 + (w / 7) * 4 + (d / 7);
        int inner = (h % 7) * 49 + (w % 7) * 7 + (d % 7);
        if (gc < 512) {
          int c = gc - 384;
          q2w[(size_t)(((c >> 5) * 64 + wdw) * NRPAD + inner) * 32 + (c & 31)] = f2bf(val);
        } else if (gc < 640) {
          int c = gc - 512;
          k2w[(size_t)(((c >> 5) * 64 + wdw) * NRPAD + inner) * 32 + (c & 31)] = f2bf(val);
        } else {
          int c = gc - 640;
          v2wT[(size_t)(((c >> 5) * 64 + wdw) * 32 + (c & 31)) * NRPAD + inner] = f2bf(val);
        }
      }
    }
  }
}

// ---------------------------------------------------------------------------
// K2: SR conv im2col (kernel==stride==4 -> disjoint patches, pure gather).
// A[p][k], k = ci*64 + kx*16 + ky*4 + kz ; matches sr_w flat layout.
// ---------------------------------------------------------------------------
__global__ void k_sr_im2col(const unsigned short* __restrict__ xbf,
                            unsigned short* __restrict__ A) {
  int idx = blockIdx.x * 256 + threadIdx.x;
  if (idx >= 343 * 16384) return;
  int p = idx >> 14, k = idx & 16383;
  int ci = k >> 6, s = k & 63;
  int kx = s >> 4, ky = (s >> 2) & 3, kz = s & 3;
  int px = p / 49, py = (p / 7) % 7, pz = p % 7;
  int t = ((px * 4 + kx) * 28 + (py * 4 + ky)) * 28 + (pz * 4 + kz);
  A[(size_t)p * 16384 + k] = xbf[(size_t)t * 256 + ci];
}

// ---------------------------------------------------------------------------
// K3: SR GEMM x1raw[343,256] = A[352,16384] @ srw^T + sr_b. grid(22,4),blk 32.
// ---------------------------------------------------------------------------
__global__ void k_gemm_sr(const unsigned short* __restrict__ A,
                          const unsigned short* __restrict__ srw,
                          const float* __restrict__ sr_b,
                          float* __restrict__ x1raw) {
  const int lane = threadIdx.x & 31;
  const int mn = lane & 15, hi = lane >> 4;
  const int mt = blockIdx.x, ng = blockIdx.y;
  v8f acc[4] = { {0,0,0,0,0,0,0,0}, {0,0,0,0,0,0,0,0},
                 {0,0,0,0,0,0,0,0}, {0,0,0,0,0,0,0,0} };
  const unsigned short* arow = A + (size_t)(mt * 16 + mn) * 16384;
  for (int ks = 0; ks < 512; ++ks) {
    __builtin_prefetch(arow + (ks + 8) * 32, 0, 0);   // global_prefetch_b8
    v16bf a = ld_frag(arow + ks * 32 + hi * 8);
    #pragma unroll
    for (int j = 0; j < 4; ++j) {
      const unsigned short* brow = srw + (size_t)(ng * 64 + j * 16 + mn) * 16384;
      v16bf b = ld_frag(brow + ks * 32 + hi * 8);
      acc[j] = wmma_bf16(a, b, acc[j]);
    }
  }
  #pragma unroll
  for (int j = 0; j < 4; ++j)
    #pragma unroll
    for (int v = 0; v < 8; ++v) {
      int t = mt * 16 + v + 8 * hi;
      int gc = ng * 64 + j * 16 + mn;
      if (t < 343) x1raw[(size_t)t * 256 + gc] = acc[j][v] + sr_b[gc];
    }
}

// ---------------------------------------------------------------------------
// K4: LayerNorm + exact GELU per row (343 rows x 256). grid 343, block 256.
// ---------------------------------------------------------------------------
__global__ void k_ln_gelu(const float* __restrict__ x1raw,
                          const float* __restrict__ g,
                          const float* __restrict__ b,
                          unsigned short* __restrict__ x1bf) {
  __shared__ float red[256];
  int row = blockIdx.x, c = threadIdx.x;
  float v = x1raw[(size_t)row * 256 + c];
  red[c] = v; __syncthreads();
  for (int s = 128; s > 0; s >>= 1) { if (c < s) red[c] += red[c + s]; __syncthreads(); }
  float mu = red[0] * (1.0f / 256.0f); __syncthreads();
  float dv = v - mu;
  red[c] = dv * dv; __syncthreads();
  for (int s = 128; s > 0; s >>= 1) { if (c < s) red[c] += red[c + s]; __syncthreads(); }
  float var = red[0] * (1.0f / 256.0f);
  float xn = dv * rsqrtf(var + 1e-5f) * g[c] + b[c];
  float ge = 0.5f * xn * (1.0f + erff(xn * 0.7071067811865475f));
  x1bf[(size_t)row * 256 + c] = f2bf(ge);
}

// ---------------------------------------------------------------------------
// K5: kv1 GEMM [352,256] @ kv1w^T -> k1 bf16 [head][352][32],
//                                    v1T bf16 [head][32][352]. grid(22,4).
// ---------------------------------------------------------------------------
__global__ void k_gemm_kv1(const unsigned short* __restrict__ x1bf,
                           const unsigned short* __restrict__ kv1w,
                           unsigned short* __restrict__ k1,
                           unsigned short* __restrict__ v1T) {
  const int lane = threadIdx.x & 31;
  const int mn = lane & 15, hi = lane >> 4;
  const int mt = blockIdx.x, ng = blockIdx.y;
  v8f acc[4] = { {0,0,0,0,0,0,0,0}, {0,0,0,0,0,0,0,0},
                 {0,0,0,0,0,0,0,0}, {0,0,0,0,0,0,0,0} };
  const unsigned short* arow = x1bf + (size_t)(mt * 16 + mn) * 256;
  for (int ks = 0; ks < 8; ++ks) {
    v16bf a = ld_frag(arow + ks * 32 + hi * 8);
    #pragma unroll
    for (int j = 0; j < 4; ++j) {
      const unsigned short* brow = kv1w + (size_t)(ng * 64 + j * 16 + mn) * 256;
      v16bf b = ld_frag(brow + ks * 32 + hi * 8);
      acc[j] = wmma_bf16(a, b, acc[j]);
    }
  }
  #pragma unroll
  for (int j = 0; j < 4; ++j)
    #pragma unroll
    for (int v = 0; v < 8; ++v) {
      int t = mt * 16 + v + 8 * hi;           // rows >=343 are harmless garbage
      int gc = ng * 64 + j * 16 + mn;         // [2][4][32]: pair, head, hd
      int head = (gc >> 5) & 3, hd = gc & 31;
      unsigned short bfv = f2bf(acc[j][v]);
      if (gc < 128) k1[(size_t)(head * NRPAD + t) * 32 + hd] = bfv;
      else          v1T[(size_t)(head * 32 + hd) * NRPAD + t] = bfv;
    }
}

// ---------------------------------------------------------------------------
// K6: LePE depthwise 3x3x3 conv (SAME). grid N, block 256 (thread=channel).
// ---------------------------------------------------------------------------
__global__ void k_lepe_conv(const float* __restrict__ y,
                            const float* __restrict__ w,
                            const float* __restrict__ bias,
                            float* __restrict__ lepe) {
  int t = blockIdx.x, c = threadIdx.x;
  int h = t / 784, ww = (t / 28) % 28, d = t % 28;
  float acc = bias[c];
  #pragma unroll
  for (int dh = -1; dh <= 1; ++dh)
    #pragma unroll
    for (int dw = -1; dw <= 1; ++dw)
      #pragma unroll
      for (int dd = -1; dd <= 1; ++dd) {
        int hh = h + dh, w2 = ww + dw, d2 = d + dd;
        if ((unsigned)hh < 28u && (unsigned)w2 < 28u && (unsigned)d2 < 28u)
          acc += y[(size_t)(((hh * 28) + w2) * 28 + d2) * 256 + c] *
                 w[c * 27 + (dh + 1) * 9 + (dw + 1) * 3 + (dd + 1)];
      }
  lepe[(size_t)t * 256 + c] = acc;
}

// ---------------------------------------------------------------------------
// K7: branch-1 attention (SR global). 1 wave per (Q-tile, head).
// S = Q(16x32) x K^T via WMMA -> LDS -> masked softmax -> P(bf16) x V via WMMA.
// grid (1372, 4), block 32.
// ---------------------------------------------------------------------------
__global__ void k_attn_sr(const unsigned short* __restrict__ q1,
                          const unsigned short* __restrict__ k1,
                          const unsigned short* __restrict__ v1T,
                          unsigned short* __restrict__ att) {
  const int lane = threadIdx.x & 31;
  const int mn = lane & 15, hi = lane >> 4;
  const int mt = blockIdx.x, h = blockIdx.y;
  __shared__ __align__(16) float sS[16 * 360];
  __shared__ __align__(16) unsigned short sP[16 * 368];
  __shared__ float sSum[16];

  const unsigned short* qrow = q1 + (size_t)(h * NTOK + mt * 16 + mn) * 32;
  v16bf aq = ld_frag(qrow + hi * 8);
  for (int tt = 0; tt < 22; ++tt) {
    const unsigned short* krow = k1 + (size_t)(h * NRPAD + tt * 16 + mn) * 32;
    v16bf bk = ld_frag(krow + hi * 8);
    v8f s = {0,0,0,0,0,0,0,0};
    s = wmma_bf16(aq, bk, s);
    #pragma unroll
    for (int v = 0; v < 8; ++v) sS[(v + 8 * hi) * 360 + tt * 16 + mn] = s[v] * SCALE_;
  }
  __syncthreads();
  if (lane < 16) {
    float* row = sS + lane * 360;
    unsigned short* prow = sP + lane * 368;
    float mx = -3.0e38f;
    for (int c = 0; c < 343; ++c) mx = fmaxf(mx, row[c]);
    float sum = 0.f;
    for (int c = 0; c < 343; ++c) { float p = __expf(row[c] - mx); sum += p; prow[c] = f2bf(p); }
    for (int c = 343; c < 352; ++c) prow[c] = 0;     // mask pad columns
    sSum[lane] = sum;
  }
  __syncthreads();
  v8f o0 = {0,0,0,0,0,0,0,0}, o1 = {0,0,0,0,0,0,0,0};
  const unsigned short* vr0 = v1T + (size_t)(h * 32 + mn) * NRPAD;
  const unsigned short* vr1 = v1T + (size_t)(h * 32 + 16 + mn) * NRPAD;
  for (int tt = 0; tt < 11; ++tt) {
    v16bf ap = ld_frag(sP + mn * 368 + tt * 32 + hi * 8);   // ds_load_b128 frags
    v16bf b0 = ld_frag(vr0 + tt * 32 + hi * 8);
    v16bf b1 = ld_frag(vr1 + tt * 32 + hi * 8);
    o0 = wmma_bf16(ap, b0, o0);
    o1 = wmma_bf16(ap, b1, o1);
  }
  #pragma unroll
  for (int v = 0; v < 8; ++v) {
    int mm = v + 8 * hi;
    int t = mt * 16 + mm;
    float inv = 1.0f / sSum[mm];
    att[(size_t)t * 256 + h * 32 + mn]      = f2bf(o0[v] * inv);
    att[(size_t)t * 256 + h * 32 + 16 + mn] = f2bf(o1[v] * inv);
  }
}

// ---------------------------------------------------------------------------
// K8: branch-2 windowed attention. grid (22 qtiles, 64 windows, 4 heads).
// Same flash structure; window-reverse scatter on store; cols 128..255.
// ---------------------------------------------------------------------------
__global__ void k_attn_win(const unsigned short* __restrict__ q2w,
                           const unsigned short* __restrict__ k2w,
                           const unsigned short* __restrict__ v2wT,
                           unsigned short* __restrict__ att) {
  const int lane = threadIdx.x & 31;
  const int mn = lane & 15, hi = lane >> 4;
  const int mt = blockIdx.x, wdw = blockIdx.y, h = blockIdx.z;
  __shared__ __align__(16) float sS[16 * 360];
  __shared__ __align__(16) unsigned short sP[16 * 368];
  __shared__ float sSum[16];

  const size_t blk = (size_t)(h * 64 + wdw);
  const unsigned short* qrow = q2w + (blk * NRPAD + mt * 16 + mn) * 32;
  v16bf aq = ld_frag(qrow + hi * 8);
  for (int tt = 0; tt < 22; ++tt) {
    const unsigned short* krow = k2w + (blk * NRPAD + tt * 16 + mn) * 32;
    v16bf bk = ld_frag(krow + hi * 8);
    v8f s = {0,0,0,0,0,0,0,0};
    s = wmma_bf16(aq, bk, s);
    #pragma unroll
    for (int v = 0; v < 8; ++v) sS[(v + 8 * hi) * 360 + tt * 16 + mn] = s[v] * SCALE_;
  }
  __syncthreads();
  if (lane < 16) {
    float* row = sS + lane * 360;
    unsigned short* prow = sP + lane * 368;
    float mx = -3.0e38f;
    for (int c = 0; c < 343; ++c) mx = fmaxf(mx, row[c]);
    float sum = 0.f;
    for (int c = 0; c < 343; ++c) { float p = __expf(row[c] - mx); sum += p; prow[c] = f2bf(p); }
    for (int c = 343; c < 352; ++c) prow[c] = 0;
    sSum[lane] = sum;
  }
  __syncthreads();
  v8f o0 = {0,0,0,0,0,0,0,0}, o1 = {0,0,0,0,0,0,0,0};
  const unsigned short* vr0 = v2wT + (blk * 32 + mn) * NRPAD;
  const unsigned short* vr1 = v2wT + (blk * 32 + 16 + mn) * NRPAD;
  for (int tt = 0; tt < 11; ++tt) {
    v16bf ap = ld_frag(sP + mn * 368 + tt * 32 + hi * 8);
    v16bf b0 = ld_frag(vr0 + tt * 32 + hi * 8);
    v16bf b1 = ld_frag(vr1 + tt * 32 + hi * 8);
    o0 = wmma_bf16(ap, b0, o0);
    o1 = wmma_bf16(ap, b1, o1);
  }
  int wx = wdw >> 4, wy = (wdw >> 2) & 3, wz = wdw & 3;
  #pragma unroll
  for (int v = 0; v < 8; ++v) {
    int inner = mt * 16 + v + 8 * hi;
    if (inner >= 343) continue;                      // pad rows: never stored
    int ih = inner / 49, iw = (inner / 7) % 7, iz = inner % 7;
    int t = ((wx * 7 + ih) * 28 + (wy * 7 + iw)) * 28 + (wz * 7 + iz);
    float inv = 1.0f / sSum[v + 8 * hi];
    att[(size_t)t * 256 + 128 + h * 32 + mn]      = f2bf(o0[v] * inv);
    att[(size_t)t * 256 + 128 + h * 32 + 16 + mn] = f2bf(o1[v] * inv);
  }
}

// ---------------------------------------------------------------------------
// K9: att (bf16) += lepe (fp32), in place (each element written once/call).
// ---------------------------------------------------------------------------
__global__ void k_add_lepe(unsigned short* __restrict__ att,
                           const float* __restrict__ lepe) {
  int i = blockIdx.x * 256 + threadIdx.x;
  if (i < NTOK * 256) att[i] = f2bf(bf2f(att[i]) + lepe[i]);
}

// ---------------------------------------------------------------------------
// K10: projection GEMM out[N,256] = att @ projw^T + proj_b (fp32 out).
// grid (1372, 4), block 32.
// ---------------------------------------------------------------------------
__global__ void k_gemm_proj(const unsigned short* __restrict__ att,
                            const unsigned short* __restrict__ projw,
                            const float* __restrict__ proj_b,
                            float* __restrict__ out) {
  const int lane = threadIdx.x & 31;
  const int mn = lane & 15, hi = lane >> 4;
  const int mt = blockIdx.x, ng = blockIdx.y;
  v8f acc[4] = { {0,0,0,0,0,0,0,0}, {0,0,0,0,0,0,0,0},
                 {0,0,0,0,0,0,0,0}, {0,0,0,0,0,0,0,0} };
  const unsigned short* arow = att + (size_t)(mt * 16 + mn) * 256;
  for (int ks = 0; ks < 8; ++ks) {
    v16bf a = ld_frag(arow + ks * 32 + hi * 8);
    #pragma unroll
    for (int j = 0; j < 4; ++j) {
      const unsigned short* brow = projw + (size_t)(ng * 64 + j * 16 + mn) * 256;
      v16bf b = ld_frag(brow + ks * 32 + hi * 8);
      acc[j] = wmma_bf16(a, b, acc[j]);
    }
  }
  #pragma unroll
  for (int j = 0; j < 4; ++j)
    #pragma unroll
    for (int v = 0; v < 8; ++v) {
      int t = mt * 16 + v + 8 * hi;
      int gc = ng * 64 + j * 16 + mn;
      out[(size_t)t * 256 + gc] = acc[j][v] + proj_b[gc];
    }
}

// ---------------------------------------------------------------------------
extern "C" void kernel_launch(void* const* d_in, const int* in_sizes, int n_in,
                              void* d_out, int out_size, void* d_ws, size_t ws_size,
                              hipStream_t stream) {
  // dict order: x,H,W,D, lepe_w, lepe_b, lepe_conv_w, lepe_conv_b, sr_w, sr_b,
  //             norm_g, norm_b, q1_w, kv1_w, q2_w, kv2_w, proj_w, proj_b
  const float* x           = (const float*)d_in[0];
  const float* lepe_w      = (const float*)d_in[4];
  const float* lepe_b      = (const float*)d_in[5];
  const float* lepe_conv_w = (const float*)d_in[6];
  const float* lepe_conv_b = (const float*)d_in[7];
  const float* sr_w        = (const float*)d_in[8];
  const float* sr_b        = (const float*)d_in[9];
  const float* norm_g      = (const float*)d_in[10];
  const float* norm_b      = (const float*)d_in[11];
  const float* q1_w        = (const float*)d_in[12];
  const float* kv1_w       = (const float*)d_in[13];
  const float* q2_w        = (const float*)d_in[14];
  const float* kv2_w       = (const float*)d_in[15];
  const float* proj_w      = (const float*)d_in[16];
  const float* proj_b      = (const float*)d_in[17];
  float* out = (float*)d_out;

  // bump-allocate workspace (~112 MB)
  char* p = (char*)d_ws;
  auto alloc = [&](size_t bytes) {
    char* r = p; p += (bytes + 255) & ~(size_t)255; return r;
  };
  unsigned short* xbf   = (unsigned short*)alloc((size_t)NTOK * 256 * 2);
  unsigned short* wcat  = (unsigned short*)alloc((size_t)768 * 256 * 2);
  unsigned short* kv1bf = (unsigned short*)alloc((size_t)256 * 256 * 2);
  unsigned short* projbf= (unsigned short*)alloc((size_t)256 * 256 * 2);
  unsigned short* srbf  = (unsigned short*)alloc((size_t)256 * 16384 * 2);
  float*          y     = (float*)alloc((size_t)NTOK * 256 * 4);
  float*          lepe  = (float*)alloc((size_t)NTOK * 256 * 4);
  unsigned short* q1    = (unsigned short*)alloc((size_t)4 * NTOK * 32 * 2);
  unsigned short* q2w   = (unsigned short*)alloc((size_t)4 * 64 * NRPAD * 32 * 2);
  unsigned short* k2w   = (unsigned short*)alloc((size_t)4 * 64 * NRPAD * 32 * 2);
  unsigned short* v2wT  = (unsigned short*)alloc((size_t)4 * 64 * NRPAD * 32 * 2);
  unsigned short* Asr   = (unsigned short*)alloc((size_t)NRPAD * 16384 * 2);
  float*          x1raw = (float*)alloc((size_t)NRPAD * 256 * 4);
  unsigned short* x1bf  = (unsigned short*)alloc((size_t)NRPAD * 256 * 2);
  unsigned short* k1    = (unsigned short*)alloc((size_t)4 * NRPAD * 32 * 2);
  unsigned short* v1T   = (unsigned short*)alloc((size_t)4 * NRPAD * 32 * 2);
  unsigned short* att   = (unsigned short*)alloc((size_t)NTOK * 256 * 2);

  k_convert_x<<<NTOK, 256, 0, stream>>>(x, xbf);
  k_pack_w<<<17664, 256, 0, stream>>>(lepe_w, q1_w, q2_w, kv2_w, kv1_w, proj_w,
                                      sr_w, wcat, kv1bf, projbf, srbf);
  k_gemm_fused<<<dim3(NT_M, 12), 32, 0, stream>>>(xbf, wcat, lepe_b, y, q1,
                                                  q2w, k2w, v2wT);
  k_sr_im2col<<<NTOK, 256, 0, stream>>>(xbf, Asr);
  k_gemm_sr<<<dim3(22, 4), 32, 0, stream>>>(Asr, srbf, sr_b, x1raw);
  k_ln_gelu<<<343, 256, 0, stream>>>(x1raw, norm_g, norm_b, x1bf);
  k_gemm_kv1<<<dim3(22, 4), 32, 0, stream>>>(x1bf, kv1bf, k1, v1T);
  k_attn_sr<<<dim3(NT_M, 4), 32, 0, stream>>>(q1, k1, v1T, att);
  k_attn_win<<<dim3(22, 64, 4), 32, 0, stream>>>(q2w, k2w, v2wT, att);
  k_lepe_conv<<<NTOK, 256, 0, stream>>>(y, lepe_conv_w, lepe_conv_b, lepe);
  k_add_lepe<<<NTOK, 256, 0, stream>>>(att, lepe);
  k_gemm_proj<<<dim3(NT_M, 4), 32, 0, stream>>>(att, projbf, proj_b, out);
  (void)in_sizes; (void)n_in; (void)out_size; (void)ws_size;
}
